// PathTree_4638564680055
// MI455X (gfx1250) — matrix operand order repaired and checked by reading, not compiled, not run on hardware
//
#include <hip/hip_runtime.h>
#include <hip/hip_bf16.h>
#include <math.h>

typedef __bf16 bf16_t;
typedef __attribute__((ext_vector_type(16))) __bf16 v16bf;
typedef __attribute__((ext_vector_type(8)))  __bf16 v8bf;
typedef __attribute__((ext_vector_type(8)))  float  v8f;

static __device__ inline v8f wmma_bf16(v16bf a, v16bf b, v8f c) {
  return __builtin_amdgcn_wmma_f32_16x16x32_bf16(
      false, a, false, b, (short)0, c, false, false);
}
static __device__ inline v16bf cat8(v8bf lo, v8bf hi) {
  return __builtin_shufflevector(lo, hi, 0, 1, 2, 3, 4, 5, 6, 7,
                                 8, 9, 10, 11, 12, 13, 14, 15);
}

// ---- fragment loaders (all contiguous vector accesses) -----------------
static __device__ inline v16bf load_a_rm(const bf16_t* base, int ld) {
  int lane = threadIdx.x & 31;
  const bf16_t* r = base + (lane & 15) * ld + ((lane >> 4) << 3);
  return cat8(*(const v8bf*)r, *(const v8bf*)(r + 16));
}
static __device__ inline v16bf load_a_f32cv(const float* base, int ld) {
  int lane = threadIdx.x & 31;
  const float* r = base + (lane & 15) * ld + ((lane >> 4) << 3);
  v8f lo = *(const v8f*)r;
  v8f hi = *(const v8f*)(r + 16);
  v16bf a;
#pragma unroll
  for (int j = 0; j < 8; ++j) { a[j] = (bf16_t)lo[j]; a[8 + j] = (bf16_t)hi[j]; }
  return a;
}
static __device__ inline v16bf load_b_swz(const bf16_t* W, int kt, int nt, int NT) {
  int lane = threadIdx.x & 31;
  return *(const v16bf*)(W + ((size_t)((kt * NT + nt) * 32 + lane) << 4));
}
// A = P^T fragment from P stored as [rowblock8][64 nodes][8]
static __device__ inline v16bf load_a_pfrag(const bf16_t* Pb, int krow0, int m0) {
  int lane = threadIdx.x & 31;
  int rb = (krow0 >> 3) + (lane >> 4);
  const bf16_t* p = Pb + (((size_t)rb * 64 + m0 + (lane & 15)) << 3);
  return cat8(*(const v8bf*)p, *(const v8bf*)(p + 2 * 64 * 8));
}
// B = x fragment from x stored as [rowblock8][512 feats][8]
static __device__ inline v16bf load_b_xfrag(const bf16_t* Xb, int krow0, int n) {
  int lane = threadIdx.x & 31;
  int rb = (krow0 >> 3) + (lane >> 4);
  const bf16_t* p = Xb + (((size_t)rb * 512 + n + (lane & 15)) << 3);
  return cat8(*(const v8bf*)p, *(const v8bf*)(p + 2 * 512 * 8));
}

// ---- prep kernels ------------------------------------------------------
__global__ void k_swz_b(const float* __restrict__ src, bf16_t* __restrict__ dst,
                        int K, int Nlog, int Nsrc) {
  int i = blockIdx.x * blockDim.x + threadIdx.x;
  if (i >= K * Nlog) return;
  int j = i & 15, lane = (i >> 4) & 31, t = i >> 9;
  int NT = Nlog >> 4;
  int nt = t % NT, kt = t / NT;
  int k = kt * 32 + ((lane >> 4) << 3) + (j & 7) + ((j >> 3) << 4);
  int n = nt * 16 + (lane & 15);
  dst[i] = (n < Nsrc) ? (bf16_t)src[(size_t)k * Nsrc + n] : (bf16_t)0.f;
}
__global__ void k_zero(float* p, int n) {
  int i = blockIdx.x * blockDim.x + threadIdx.x;
  if (i < n) p[i] = 0.f;
}
__global__ void k_pad_node_feat(const float* __restrict__ nf, bf16_t* __restrict__ d) {
  int i = blockIdx.x * blockDim.x + threadIdx.x;          // 64*512
  if (i >= 64 * 512) return;
  d[i] = (i < 63 * 512) ? (bf16_t)nf[i] : (bf16_t)0.f;
}

// ---- fused pass: x = leaky(patch@W1+b); gate; P = exp(gate@Wattn+b) ----
__global__ void k_main(const float* __restrict__ patch,
                       const bf16_t* __restrict__ w1s, const float* __restrict__ b1,
                       const bf16_t* __restrict__ vas, const float* __restrict__ vab,
                       const bf16_t* __restrict__ uas, const float* __restrict__ uab,
                       const bf16_t* __restrict__ wats, const float* __restrict__ watb,
                       bf16_t* __restrict__ Xb, bf16_t* __restrict__ Pb,
                       float* __restrict__ colsum, int N) {
  __shared__ bf16_t xt[2][16][520];   // per-wave 16x512 x tile (padded ld)
  __shared__ bf16_t gt[2][16][264];   // per-wave 16x256 gate tile
  int wave = threadIdx.x >> 5;
  int tile = blockIdx.x * 2 + wave;
  if (tile >= (N >> 4)) return;
  int lane = threadIdx.x & 31;
  int col_l = lane & 15, mb = (lane >> 4) << 3;
  int rb = tile * 2 + (lane >> 4);    // row-block-8 index this lane stores
  const float* Ab = patch + (size_t)tile * 16 * 512;

  // phase 1: x  (output in 4 chunks of 128 cols, acc[8])
  for (int nc = 0; nc < 4; ++nc) {
    v8f acc[8] = {};
    for (int k = 0; k < 16; ++k) {
      v16bf a = load_a_f32cv(Ab + k * 32, 512);
#pragma unroll
      for (int j = 0; j < 8; ++j)
        acc[j] = wmma_bf16(a, load_b_swz(w1s, k, nc * 8 + j, 32), acc[j]);
    }
#pragma unroll
    for (int j = 0; j < 8; ++j) {
      int col = nc * 128 + j * 16 + col_l;
      float bias = b1[col];
      v8bf pk;
#pragma unroll
      for (int e = 0; e < 8; ++e) {
        float v = acc[j][e] + bias;
        v = (v >= 0.f) ? v : 0.01f * v;
        bf16_t bv = (bf16_t)v;
        pk[e] = bv;
        xt[wave][mb + e][col] = bv;
      }
      *(v8bf*)(Xb + (((size_t)rb * 512 + col) << 3)) = pk;   // 16B store
    }
  }

  // phase 2: gate = tanh(x@Va+b) * sigmoid(x@Ua+b)
  for (int nc = 0; nc < 4; ++nc) {
    v8f aV[4] = {}, aU[4] = {};
    for (int k = 0; k < 16; ++k) {
      v16bf a = load_a_rm(&xt[wave][0][0] + k * 32, 520);
#pragma unroll
      for (int j = 0; j < 4; ++j) {
        aV[j] = wmma_bf16(a, load_b_swz(vas, k, nc * 4 + j, 16), aV[j]);
        aU[j] = wmma_bf16(a, load_b_swz(uas, k, nc * 4 + j, 16), aU[j]);
      }
    }
#pragma unroll
    for (int j = 0; j < 4; ++j) {
      int col = nc * 64 + j * 16 + col_l;
      float bv = vab[col], bu = uab[col];
#pragma unroll
      for (int e = 0; e < 8; ++e) {
        float gv = tanhf(aV[j][e] + bv) * (1.f / (1.f + __expf(-(aU[j][e] + bu))));
        gt[wave][mb + e][col] = (bf16_t)gv;
      }
    }
  }

  // phase 3: P = exp(gate @ Wattn + b); column sums for softmax denominator
  v8f acc[4] = {};
  for (int k = 0; k < 8; ++k) {
    v16bf a = load_a_rm(&gt[wave][0][0] + k * 32, 264);
#pragma unroll
    for (int j = 0; j < 4; ++j)
      acc[j] = wmma_bf16(a, load_b_swz(wats, k, j, 4), acc[j]);
  }
#pragma unroll
  for (int j = 0; j < 4; ++j) {
    int col = j * 16 + col_l;
    float bias = (col < 63) ? watb[col] : 0.f;
    float part = 0.f;
    v8bf pk;
#pragma unroll
    for (int e = 0; e < 8; ++e) {
      float p = __expf(acc[j][e] + bias);
      part += p;
      pk[e] = (bf16_t)p;
    }
    *(v8bf*)(Pb + (((size_t)rb * 64 + col) << 3)) = pk;      // 16B store
    atomicAdd(&colsum[col], part);
  }
}

// ---- pass 3: ts(64x512) += P^T @ x, split-K, M=32 per wave -------------
__global__ void k_tree_slide(const bf16_t* __restrict__ Pb, const bf16_t* __restrict__ Xb,
                             float* __restrict__ ts, int N, int kc_steps) {
  int wave = threadIdx.x >> 5;                  // 8 jobs: 2 M-strips x 4 N-chunks
  int m0 = (wave & 1) << 5;
  int n0 = (wave >> 1) << 7;
  int lane = threadIdx.x & 31;
  int col_l = lane & 15, mb = (lane >> 4) << 3;
  int n32 = N >> 5;
  int k0 = blockIdx.x * kc_steps;
  int k1 = k0 + kc_steps; if (k1 > n32) k1 = n32;
  v8f acc[2][8] = {};
  for (int ks = k0; ks < k1; ++ks) {
    int krow = ks << 5;
    v16bf a0 = load_a_pfrag(Pb, krow, m0);
    v16bf a1 = load_a_pfrag(Pb, krow, m0 + 16);
#pragma unroll
    for (int j = 0; j < 8; ++j) {
      v16bf b = load_b_xfrag(Xb, krow, n0 + j * 16);
      acc[0][j] = wmma_bf16(a0, b, acc[0][j]);
      acc[1][j] = wmma_bf16(a1, b, acc[1][j]);
    }
  }
#pragma unroll
  for (int s = 0; s < 2; ++s)
#pragma unroll
    for (int j = 0; j < 8; ++j) {
      int col = n0 + j * 16 + col_l;
#pragma unroll
      for (int e = 0; e < 8; ++e)
        atomicAdd(&ts[(size_t)(m0 + s * 16 + mb + e) * 512 + col], acc[s][j][e]);
    }
}

// ---- node-side GEMM: out = leaky(A(64x512) @ W(512x512) + bias), f32 ---
__global__ void k_node_gemm(const bf16_t* __restrict__ A, const bf16_t* __restrict__ W,
                            const float* __restrict__ bias, float* __restrict__ out) {
  int wave = threadIdx.x >> 5;
  int tid = blockIdx.x * 4 + wave;              // 128 tiles: 4 M x 32 N
  int m0 = (tid & 3) << 4;
  int nt = tid >> 2;
  int lane = threadIdx.x & 31;
  v8f acc = {};
  for (int k = 0; k < 16; ++k) {
    v16bf a = load_a_rm(A + m0 * 512 + k * 32, 512);
    acc = wmma_bf16(a, load_b_swz(W, k, nt, 32), acc);
  }
  int col = (nt << 4) + (lane & 15);
  int mb = (lane >> 4) << 3;
  float bv = bias[col];
#pragma unroll
  for (int e = 0; e < 8; ++e) {
    float v = acc[e] + bv;
    v = (v >= 0.f) ? v : 0.01f * v;
    out[(m0 + mb + e) * 512 + col] = v;
  }
}

// ---- tree messages -----------------------------------------------------
__global__ void k_msg_parent(const float* __restrict__ t, const int* __restrict__ e,
                             int nedge, bf16_t* __restrict__ s1) {
  int i = blockIdx.x * blockDim.x + threadIdx.x;
  if (i >= 64 * 512) return;
  int node = i >> 9, d = i & 511;
  float v = 0.f;
  if (node < 63) {
    v = t[i];
    const int* src = e;              // parents
    const int* dst = e + nedge;      // children
    for (int k = 0; k < nedge; ++k)
      if (dst[k] == node) { v += t[src[k] * 512 + d]; break; }
  }
  s1[i] = (bf16_t)v;
}
__global__ void k_msg_children(const float* __restrict__ h, const int* __restrict__ e,
                               int nedge, bf16_t* __restrict__ s2) {
  int i = blockIdx.x * blockDim.x + threadIdx.x;
  if (i >= 64 * 512) return;
  int node = i >> 9, d = i & 511;
  float v = 0.f;
  if (node < 63) {
    v = h[i];
    const int* src = e;              // children
    const int* dst = e + nedge;      // parents
    for (int k = 0; k < nedge; ++k)
      if (dst[k] == node) v += h[src[k] * 512 + d];
  }
  s2[i] = (bf16_t)v;
}

// ---- finalize: softmax normalize, tree walk, cosine logits -------------
__global__ void k_finalize(float* __restrict__ ts, const float* __restrict__ colsum,
                           const float* __restrict__ tt, const float* __restrict__ lscale,
                           float* __restrict__ f, float* __restrict__ outp) {
  __shared__ float cs[64];
  __shared__ float red[256];
  __shared__ float snorm;
  int tid = threadIdx.x;
  for (int i = tid; i < 63 * 512; i += blockDim.x) ts[i] = ts[i] / colsum[i >> 9];
  __syncthreads();
  if (tid < 63) {
    float s = 0.f;
    for (int d = 0; d < 512; ++d) s += ts[tid * 512 + d] * tt[tid * 512 + d];
    cs[tid] = s;
  }
  __syncthreads();
  for (int i = tid; i < 32 * 512; i += blockDim.x) f[i] = ts[i];
  __syncthreads();
  const int lvl[7] = {0, 32, 48, 56, 60, 62, 63};
  for (int li = 1; li < 6; ++li) {
    int s0 = lvl[li], s1 = lvl[li + 1], prev = lvl[li - 1];
    for (int p = s0; p < s1; ++p) {
      int c1 = prev + 2 * (p - s0), c2 = c1 + 1;
      float e2 = __expf(cs[c2]), e1 = __expf(cs[c1]);
      float p0 = e2 / (e2 + e1), p1 = 1.f - p0;
      for (int d = tid; d < 512; d += blockDim.x)
        f[p * 512 + d] = ts[p * 512 + d] + p0 * f[c2 * 512 + d] + p1 * f[c1 * 512 + d];
    }
  }
  __syncthreads();
  float part = 0.f;
  for (int d = tid; d < 512; d += blockDim.x) { float v = f[62 * 512 + d]; part += v * v; }
  red[tid] = part;
  __syncthreads();
  for (int s = 128; s > 0; s >>= 1) { if (tid < s) red[tid] += red[tid + s]; __syncthreads(); }
  if (tid == 0) snorm = sqrtf(red[0]);
  __syncthreads();
  if (tid < 32) {
    float ln = 0.f, dot = 0.f;
    for (int d = 0; d < 512; ++d) {
      float w = tt[tid * 512 + d];
      ln += w * w;
      dot += f[62 * 512 + d] * w;
    }
    outp[tid] = __expf(lscale[0]) * dot / (snorm * sqrtf(ln));
  }
}

extern "C" void kernel_launch(void* const* d_in, const int* in_sizes, int n_in,
                              void* d_out, int out_size, void* d_ws, size_t ws_size,
                              hipStream_t stream) {
  const float* patch  = (const float*)d_in[0];
  const float* nf     = (const float*)d_in[1];
  const float* pfw    = (const float*)d_in[2];
  const float* pfb    = (const float*)d_in[3];
  const float* tfw    = (const float*)d_in[4];
  const float* tfb    = (const float*)d_in[5];
  const float* Va     = (const float*)d_in[6];
  const float* Vab    = (const float*)d_in[7];
  const float* Ua     = (const float*)d_in[8];
  const float* Uab    = (const float*)d_in[9];
  const float* Wat    = (const float*)d_in[10];
  const float* Watb   = (const float*)d_in[11];
  const float* Wd     = (const float*)d_in[12];
  const float* bd     = (const float*)d_in[13];
  const float* Wu     = (const float*)d_in[14];
  const float* bu     = (const float*)d_in[15];
  const float* lscale = (const float*)d_in[16];
  const int*   up2dn  = (const int*)d_in[17];
  const int*   dn2up  = (const int*)d_in[18];
  const int N = in_sizes[0] / 512;
  const int nedge = in_sizes[17] / 2;

  char* w = (char*)d_ws;
  auto take = [&](size_t bytes) { void* p = (void*)w; w += (bytes + 255) & ~(size_t)255; return p; };
  bf16_t* Xb   = (bf16_t*)take((size_t)N * 512 * 2);   // [rb8][512][8]
  bf16_t* Pb   = (bf16_t*)take((size_t)N * 64 * 2);    // [rb8][64][8]
  bf16_t* w1s  = (bf16_t*)take(512 * 512 * 2);
  bf16_t* vas  = (bf16_t*)take(512 * 256 * 2);
  bf16_t* uas  = (bf16_t*)take(512 * 256 * 2);
  bf16_t* wats = (bf16_t*)take(256 * 64 * 2);
  bf16_t* nfb  = (bf16_t*)take(64 * 512 * 2);
  bf16_t* wts  = (bf16_t*)take(512 * 512 * 2);
  bf16_t* wds  = (bf16_t*)take(512 * 512 * 2);
  bf16_t* wus  = (bf16_t*)take(512 * 512 * 2);
  float*  csum = (float*)take(64 * 4);
  float*  ts   = (float*)take(64 * 512 * 4);
  float*  t_   = (float*)take(64 * 512 * 4);
  bf16_t* s1b  = (bf16_t*)take(64 * 512 * 2);
  float*  h_   = (float*)take(64 * 512 * 4);
  bf16_t* s2b  = (bf16_t*)take(64 * 512 * 2);
  float*  tt   = (float*)take(64 * 512 * 4);
  float*  f_   = (float*)take(64 * 512 * 4);

  // prep: swizzle weights into fragment-major bf16 + zero accumulators
  k_swz_b<<<(512 * 512 + 255) / 256, 256, 0, stream>>>(pfw, w1s, 512, 512, 512);
  k_swz_b<<<(512 * 256 + 255) / 256, 256, 0, stream>>>(Va, vas, 512, 256, 256);
  k_swz_b<<<(512 * 256 + 255) / 256, 256, 0, stream>>>(Ua, uas, 512, 256, 256);
  k_swz_b<<<(256 * 64 + 255) / 256, 256, 0, stream>>>(Wat, wats, 256, 64, 63);
  k_swz_b<<<(512 * 512 + 255) / 256, 256, 0, stream>>>(tfw, wts, 512, 512, 512);
  k_swz_b<<<(512 * 512 + 255) / 256, 256, 0, stream>>>(Wd, wds, 512, 512, 512);
  k_swz_b<<<(512 * 512 + 255) / 256, 256, 0, stream>>>(Wu, wus, 512, 512, 512);
  k_pad_node_feat<<<(64 * 512 + 255) / 256, 256, 0, stream>>>(nf, nfb);
  k_zero<<<1, 64, 0, stream>>>(csum, 64);
  k_zero<<<(64 * 512 + 255) / 256, 256, 0, stream>>>(ts, 64 * 512);

  // fused patch pipeline
  int ntiles = N >> 4;
  k_main<<<(ntiles + 1) / 2, 64, 0, stream>>>(patch, w1s, pfb, vas, Vab, uas, Uab,
                                              wats, Watb, Xb, Pb, csum, N);
  // split-K reduction  ts += P^T @ x
  int n32 = N >> 5;
  int kc = (n32 + 24) / 25;
  k_tree_slide<<<dim3(25), 256, 0, stream>>>(Pb, Xb, ts, N, kc);

  // node/tree side
  k_node_gemm<<<32, 128, 0, stream>>>(nfb, wts, tfb, t_);
  k_msg_parent<<<(64 * 512 + 255) / 256, 256, 0, stream>>>(t_, up2dn, nedge, s1b);
  k_node_gemm<<<32, 128, 0, stream>>>(s1b, wds, bd, h_);
  k_msg_children<<<(64 * 512 + 255) / 256, 256, 0, stream>>>(h_, dn2up, nedge, s2b);
  k_node_gemm<<<32, 128, 0, stream>>>(s2b, wus, bu, tt);

  k_finalize<<<1, 256, 0, stream>>>(ts, csum, tt, lscale, f_, (float*)d_out);
}
